// MoEBlock_74105365725772
// MI455X (gfx1250) — matrix operand order repaired and checked by reading, not compiled, not run on hardware
//
#include <hip/hip_runtime.h>
#include <math.h>

// ---------------- problem constants ----------------
constexpr int TOK = 4 * 4096;     // B*S tokens
constexpr int DD  = 1024;         // model dim
constexpr int EE  = 8;            // experts
constexpr int HH  = 384;          // hidden dim
constexpr int CAP = TOK;          // per-expert token-list capacity
// x-tile LDS layout matches TDM pad config: 16B pad inserted after every 1024B
// -> row stride 2080B (1040 bf16), 16B gap after element 512 within a row.
constexpr int XSROW = 1040;       // bf16 row stride for x tile
constexpr int HSD   = 384 + 8;    // LDS stride (bf16) for hidden tile rows
constexpr int FFN_BLOCKS = 2 * TOK / 16 + EE;  // upper bound on 16-token tiles

typedef __attribute__((ext_vector_type(16))) __bf16 v16bf;
typedef __attribute__((ext_vector_type(8)))  __bf16 bf16x8;
typedef __attribute__((ext_vector_type(8)))  float  v8f;
typedef __attribute__((ext_vector_type(4)))  unsigned int u32x4;
typedef __attribute__((ext_vector_type(8)))  int i32x8;
typedef __attribute__((ext_vector_type(4)))  int i32x4;

// TDM gather path. Arity differs by toolchain (bridge doc):
//   ROCm 7.2 / clang-22:       5 args (u32x4, i32x8, i32x4, i32x4, i32)
//   amdgpu-toolchain/clang-23: 6 args (u32x4, i32x8, i32x4, i32x4, i32x8, i32)
// The therock lane is identified by its TDM header being present.
#if defined(__gfx1250__) && __has_builtin(__builtin_amdgcn_tensor_load_to_lds)
#  if __has_include(<hip/amd_detail/amd_gfx1250_TDM.h>)
#    define TDM_MODE 6
#  else
#    define TDM_MODE 5
#  endif
#else
#  define TDM_MODE 0
#endif

__device__ __forceinline__ v16bf cat8(bf16x8 lo, bf16x8 hi) {
    return __builtin_shufflevector(lo, hi, 0,1,2,3,4,5,6,7,8,9,10,11,12,13,14,15);
}
__device__ __forceinline__ float gelu_exact(float v) {
    return 0.5f * v * (1.0f + erff(v * 0.70710678118654752440f));
}

// ---------------- zero output ----------------
__global__ void zero_kernel(float* __restrict__ p, size_t n) {
    size_t i = (size_t)blockIdx.x * blockDim.x + threadIdx.x;
    if (i < n) p[i] = 0.0f;
}

// ---------------- fp32 -> bf16 convert (x) ----------------
__global__ __launch_bounds__(256) void conv_x_kernel(const float* __restrict__ x,
                                                     __bf16* __restrict__ xb) {
    size_t i = ((size_t)blockIdx.x * 256 + threadIdx.x) * 8;
    float4 a = *(const float4*)(x + i);
    float4 b = *(const float4*)(x + i + 4);
    bf16x8 o = { (__bf16)a.x, (__bf16)a.y, (__bf16)a.z, (__bf16)a.w,
                 (__bf16)b.x, (__bf16)b.y, (__bf16)b.z, (__bf16)b.w };
    *(bf16x8*)(xb + i) = o;
}

// ---------------- weight transpose + convert: w[e][A][Bd] -> wt[e][Bd][A] (bf16) ----------------
__global__ __launch_bounds__(256) void transpose_w_kernel(const float* __restrict__ w,
                                                          __bf16* __restrict__ wt,
                                                          int Adim, int Bdim) {
    __shared__ float t[32][33];
    const float* wp = w  + (size_t)blockIdx.z * Adim * Bdim;
    __bf16*      wo = wt + (size_t)blockIdx.z * Adim * Bdim;
    int a0 = blockIdx.x * 32, b0 = blockIdx.y * 32;
    #pragma unroll
    for (int p = 0; p < 4; ++p) {
        int r = p * 8 + (threadIdx.x >> 5);
        int c = threadIdx.x & 31;
        t[r][c] = wp[(size_t)(a0 + r) * Bdim + (b0 + c)];
    }
    __syncthreads();
    #pragma unroll
    for (int p = 0; p < 4; ++p) {
        int r = p * 8 + (threadIdx.x >> 5);
        int c = threadIdx.x & 31;
        wo[(size_t)(b0 + r) * Adim + (a0 + c)] = (__bf16)t[c][r];
    }
}

// ---------------- gating: logits, softmax, top2, renorm; deterministic partials ----------------
__global__ __launch_bounds__(256) void gate_kernel(
    const float* __restrict__ x, const float* __restrict__ gw,
    const float* __restrict__ gb, int* __restrict__ tidx,
    float* __restrict__ tw, float* __restrict__ pf, float* __restrict__ pc)
{
    __shared__ float sgw[DD * EE];     // 32 KB: gate_w staged
    __shared__ float sx[256 * 33];     // 33 KB: 256 tokens x 32-wide chunk (padded)
    __shared__ float sred[256 * EE];   // 8 KB: reduction buffer
    const int tid = threadIdx.x;
    for (int i = tid; i < DD * EE; i += 256) sgw[i] = gw[i];

    float acc[EE];
    #pragma unroll
    for (int e = 0; e < EE; ++e) acc[e] = gb[e];

    const int token0 = blockIdx.x * 256;
    for (int kc = 0; kc < DD; kc += 32) {
        __syncthreads();
        for (int i = tid; i < 256 * 32; i += 256) {
            int row = i >> 5, c = i & 31;
            sx[row * 33 + c] = x[(size_t)(token0 + row) * DD + kc + c];
        }
        __syncthreads();
        #pragma unroll 8
        for (int k = 0; k < 32; ++k) {
            float xv = sx[tid * 33 + k];
            const float* g = &sgw[(kc + k) * EE];
            #pragma unroll
            for (int e = 0; e < EE; ++e) acc[e] = fmaf(xv, g[e], acc[e]);
        }
    }
    // softmax over 8
    float mx = acc[0];
    #pragma unroll
    for (int e = 1; e < EE; ++e) mx = fmaxf(mx, acc[e]);
    float p[EE], sum = 0.0f;
    #pragma unroll
    for (int e = 0; e < EE; ++e) { p[e] = expf(acc[e] - mx); sum += p[e]; }
    float inv = 1.0f / sum;
    #pragma unroll
    for (int e = 0; e < EE; ++e) p[e] *= inv;
    // top-2 (ties -> lower index, matching jax.lax.top_k)
    int i0 = 0; float v0 = p[0];
    #pragma unroll
    for (int e = 1; e < EE; ++e) if (p[e] > v0) { v0 = p[e]; i0 = e; }
    int i1 = (i0 == 0) ? 1 : 0; float v1 = p[i1];
    #pragma unroll
    for (int e = 0; e < EE; ++e) if (e != i0 && p[e] > v1) { v1 = p[e]; i1 = e; }
    float s2 = v0 + v1 + 1e-9f;
    int token = token0 + tid;
    tidx[2 * token] = i0; tidx[2 * token + 1] = i1;
    tw[2 * token] = v0 / s2; tw[2 * token + 1] = v1 / s2;

    // deterministic per-block sums: gate-prob sums (load loss) and routing counts
    __syncthreads();
    #pragma unroll
    for (int e = 0; e < EE; ++e) sred[tid * EE + e] = p[e];
    __syncthreads();
    if (tid < EE) {
        float s = 0.0f;
        for (int i = 0; i < 256; ++i) s += sred[i * EE + tid];
        pf[blockIdx.x * EE + tid] = s;
    }
    __syncthreads();
    #pragma unroll
    for (int e = 0; e < EE; ++e) sred[tid * EE + e] = (e == i0 || e == i1) ? 1.0f : 0.0f;
    __syncthreads();
    if (tid < EE) {
        float s = 0.0f;
        for (int i = 0; i < 256; ++i) s += sred[i * EE + tid];
        pc[blockIdx.x * EE + tid] = s;
    }
}

// ---------------- deterministic per-expert token lists (block scan) ----------------
__global__ __launch_bounds__(256) void route_kernel(const int* __restrict__ tidx,
                                                    int* __restrict__ list) {
    const int e = blockIdx.x;
    __shared__ int s[256];
    const int tid = threadIdx.x;
    int base = 0;
    for (int start = 0; start < TOK; start += 256) {
        int token = start + tid;
        int pred = (tidx[2 * token] == e || tidx[2 * token + 1] == e) ? 1 : 0;
        s[tid] = pred; __syncthreads();
        for (int off = 1; off < 256; off <<= 1) {
            int v = (tid >= off) ? s[tid - off] : 0;
            __syncthreads();
            s[tid] += v;
            __syncthreads();
        }
        int rank = s[tid] - pred;
        if (pred) list[e * CAP + base + rank] = token;
        int tot = s[255];
        __syncthreads();
        base += tot;
    }
    int pad = (16 - (base & 15)) & 15;
    if (tid < pad) list[e * CAP + base + tid] = -1;  // padded sentinel rows
}

// ---------------- finalize: load loss + tile offsets ----------------
__global__ void finalize_kernel(const float* __restrict__ pf, const float* __restrict__ pc,
                                int* __restrict__ toff, float* __restrict__ out_loss) {
    if (threadIdx.x == 0 && blockIdx.x == 0) {
        float loss = 0.0f;
        int off = 0;
        toff[0] = 0;
        for (int e = 0; e < EE; ++e) {
            float s = 0.0f, c = 0.0f;
            for (int b = 0; b < 64; ++b) { s += pf[b * EE + e]; c += pc[b * EE + e]; }
            float mean = s / (float)TOK;
            float d = mean - (1.0f / (float)EE);
            loss += d * d;
            int n = (int)(c + 0.5f);
            off += (n + 15) >> 4;
            toff[e + 1] = off;
        }
        out_loss[0] = loss / (float)EE;
    }
}

// x-tile LDS addressing helper: TDM pad = 16B (8 bf16) after each 512 elements
__device__ __forceinline__ int xrow_off(int row, int col) {
    return row * XSROW + col + ((col >= 512) ? 8 : 0);
}

// ---------------- fused expert FFN: 16 gathered tokens per block, bf16 WMMA ----------------
__global__ __launch_bounds__(256) void ffn_kernel(
    const __bf16* __restrict__ xb,  const __bf16* __restrict__ w1t,
    const __bf16* __restrict__ w2t, const float* __restrict__ b1,
    const float* __restrict__ b2,   const int* __restrict__ tidx,
    const float* __restrict__ tw,   const int* __restrict__ toff,
    const int* __restrict__ list,   float* __restrict__ out)
{
    __shared__ __bf16 xs[16 * XSROW];  // 32.5 KB: gathered x tile, bf16 (TDM layout)
    __shared__ __bf16 hs[16 * HSD];    // 12.25 KB: hidden tile, bf16
    __shared__ int toks[16];

    const int blk = blockIdx.x;
    if (blk >= toff[EE]) return;
    int e = 0;
    while (e < EE - 1 && blk >= toff[e + 1]) ++e;
    const int t = blk - toff[e];

    const int tid = threadIdx.x;
    if (tid < 16) toks[tid] = list[e * CAP + t * 16 + tid];
    __syncthreads();

#if TDM_MODE
    // Tensor Data Mover, gather mode: DMA 16 token rows (1024 bf16 each) from
    // global into LDS with row indices = token ids; 16B LDS pad per 1024B.
    if (tid < 32) {
        int tks[16];
        #pragma unroll
        for (int r = 0; r < 16; ++r) { int tk = toks[r]; tks[r] = (tk >= 0) ? tk : 0; }
        unsigned long long ga = (unsigned long long)(uintptr_t)xb;
        unsigned int       la = (unsigned int)(uintptr_t)(&xs[0]);
        // D# group0: count=1 | gather_mode=1 (16-bit idx) | lds_addr | global_addr | type=2
        u32x4 g0 = { 0x80000001u,
                     la,
                     (unsigned int)ga,
                     (unsigned int)((ga >> 32) & 0x01FFFFFFu) | (2u << 30) };
        // D# group1: data_size=2B, pad 4 dwords per 256 dwords, dims/strides, 16 indices
        // dword0: data_size(1)<<16 | pad_en<<20 | pad_interval(7)<<22 | pad_amount(3)<<25
        i32x8 g1 = { (int)0x07D10000u,
                     (int)((unsigned)DD << 16),          // tensor_dim0 lo16 << 16
                     (int)((unsigned)TOK << 16),         // dim0 hi (0) | tensor_dim1 lo16
                     (int)((unsigned)DD << 16),          // dim1 hi (0) | tile_dim0
                     16,                                 // tile_dim1 = #indices
                     DD,                                 // tensor_dim0_stride lo32
                     0, 0 };
        // D# groups 2/3: sixteen 16-bit row indices (token ids, ascending)
        i32x4 g2 = { tks[0] | (tks[1] << 16), tks[2] | (tks[3] << 16),
                     tks[4] | (tks[5] << 16), tks[6] | (tks[7] << 16) };
        i32x4 g3 = { tks[8]  | (tks[9]  << 16), tks[10] | (tks[11] << 16),
                     tks[12] | (tks[13] << 16), tks[14] | (tks[15] << 16) };
#if TDM_MODE == 6
        i32x8 gz = { 0, 0, 0, 0, 0, 0, 0, 0 };
        __builtin_amdgcn_tensor_load_to_lds(g0, g1, g2, g3, gz, 0);
#else
        __builtin_amdgcn_tensor_load_to_lds(g0, g1, g2, g3, 0);
#endif
        __builtin_amdgcn_s_wait_tensorcnt((short)0);
    }
    __syncthreads();
#else
    // manual gather fallback (same padded LDS layout)
    for (int c = tid; c < 16 * 128; c += 256) {
        int row = c >> 7;
        int col = (c & 127) << 3;
        int tok = toks[row];
        bf16x8 v = {};
        if (tok >= 0) v = *(const bf16x8*)(xb + (size_t)tok * DD + col);
        *(bf16x8*)(&xs[xrow_off(row, col)]) = v;
    }
    __syncthreads();
#endif

    const int lane  = tid & 31;
    const int wid   = tid >> 5;
    const int mrow  = lane & 15;
    const int hi    = lane >> 4;
    const int kselA = hi * 16;   // A-frag: lanes 16-31 take K+16..K+31
    const int kselB = hi * 16;   // B-frag: lanes 16-31 take K+16..K+31
    const int mbase = hi * 8;    // C/D-frag: lanes 16-31 hold rows 8..15

    // ---- GEMM1: hidden[16 x 384] = gelu(x @ w1 + b1); 3 H-tiles per wave ----
    v8f acc0 = {}, acc1 = {}, acc2 = {};
    const int n0 = (wid * 3 + 0) * 16 + mrow;
    const int n1 = (wid * 3 + 1) * 16 + mrow;
    const int n2 = (wid * 3 + 2) * 16 + mrow;
    const __bf16* w1e = w1t + (size_t)e * HH * DD;   // [H][D], K-contiguous
    const __bf16* pb0 = w1e + (size_t)n0 * DD + kselB;
    const __bf16* pb1 = w1e + (size_t)n1 * DD + kselB;
    const __bf16* pb2 = w1e + (size_t)n2 * DD + kselB;
    for (int kc = 0; kc < DD; kc += 32) {
        const int abase = xrow_off(mrow, kc + kselA);
        bf16x8 alo = *(const bf16x8*)(&xs[abase]);
        bf16x8 ahi = *(const bf16x8*)(&xs[abase + 8]);
        v16bf A = cat8(alo, ahi);
        v16bf B0 = cat8(*(const bf16x8*)(pb0 + kc), *(const bf16x8*)(pb0 + kc + 8));
        acc0 = __builtin_amdgcn_wmma_f32_16x16x32_bf16(false, A, false, B0, (short)0, acc0, false, false);
        v16bf B1 = cat8(*(const bf16x8*)(pb1 + kc), *(const bf16x8*)(pb1 + kc + 8));
        acc1 = __builtin_amdgcn_wmma_f32_16x16x32_bf16(false, A, false, B1, (short)0, acc1, false, false);
        v16bf B2 = cat8(*(const bf16x8*)(pb2 + kc), *(const bf16x8*)(pb2 + kc + 8));
        acc2 = __builtin_amdgcn_wmma_f32_16x16x32_bf16(false, A, false, B2, (short)0, acc2, false, false);
        __builtin_prefetch(pb0 + kc + 64, 0, 1);
    }
    {
        const float bb0 = b1[e * HH + n0];
        const float bb1 = b1[e * HH + n1];
        const float bb2 = b1[e * HH + n2];
        #pragma unroll
        for (int r = 0; r < 8; ++r) {
            hs[(mbase + r) * HSD + n0] = (__bf16)gelu_exact(acc0[r] + bb0);
            hs[(mbase + r) * HSD + n1] = (__bf16)gelu_exact(acc1[r] + bb1);
            hs[(mbase + r) * HSD + n2] = (__bf16)gelu_exact(acc2[r] + bb2);
        }
    }
    __syncthreads();

    // ---- GEMM2: out_e[16 x 1024] = hidden @ w2 + b2; 8 D-tiles per wave ----
    v8f o[8] = {};
    const __bf16* w2e = w2t + (size_t)e * DD * HH;   // [D][H], K-contiguous
    for (int kc = 0; kc < HH; kc += 32) {
        bf16x8 alo = *(const bf16x8*)(&hs[mrow * HSD + kc + kselA]);
        bf16x8 ahi = *(const bf16x8*)(&hs[mrow * HSD + kc + 8 + kselA]);
        v16bf A = cat8(alo, ahi);
        #pragma unroll
        for (int j = 0; j < 8; ++j) {
            int nc = (wid * 8 + j) * 16 + mrow;
            const __bf16* p = w2e + (size_t)nc * HH + kc + kselB;
            v16bf Bf = cat8(*(const bf16x8*)p, *(const bf16x8*)(p + 8));
            o[j] = __builtin_amdgcn_wmma_f32_16x16x32_bf16(false, A, false, Bf, (short)0, o[j], false, false);
        }
    }

    float bcol[8];
    #pragma unroll
    for (int j = 0; j < 8; ++j) bcol[j] = b2[e * DD + (wid * 8 + j) * 16 + mrow];

    // weighted combine: exactly 2 commutative f32 atomic adds per output element
    #pragma unroll
    for (int r = 0; r < 8; ++r) {
        int tok = toks[mbase + r];
        if (tok < 0) continue;
        float w = (tidx[2 * tok] == e) ? tw[2 * tok] : tw[2 * tok + 1];
        float* orow = out + (size_t)tok * DD;
        #pragma unroll
        for (int j = 0; j < 8; ++j) {
            int nc = (wid * 8 + j) * 16 + mrow;
            atomicAdd(orow + nc, w * (o[j][r] + bcol[j]));
        }
    }
}

// ---------------- launch ----------------
extern "C" void kernel_launch(void* const* d_in, const int* in_sizes, int n_in,
                              void* d_out, int out_size, void* d_ws, size_t ws_size,
                              hipStream_t stream) {
    (void)in_sizes; (void)n_in; (void)ws_size;
    const float* x  = (const float*)d_in[0];
    const float* gw = (const float*)d_in[1];
    const float* gb = (const float*)d_in[2];
    const float* w1 = (const float*)d_in[3];
    const float* b1 = (const float*)d_in[4];
    const float* w2 = (const float*)d_in[5];
    const float* b2 = (const float*)d_in[6];
    float* out = (float*)d_out;

    // workspace layout (bytes, 256-aligned sections)
    char* ws = (char*)d_ws;
    constexpr size_t OFF_XB   = 0;                                   // bf16 [TOK][D]   33.5 MB
    constexpr size_t OFF_W1T  = OFF_XB  + (size_t)TOK * DD * 2;      // bf16 [E][H][D]   6.3 MB
    constexpr size_t OFF_W2T  = OFF_W1T + (size_t)EE * HH * DD * 2;  // bf16 [E][D][H]   6.3 MB
    constexpr size_t OFF_TIDX = OFF_W2T + (size_t)EE * DD * HH * 2;  // int  [TOK][2]
    constexpr size_t OFF_TW   = OFF_TIDX + (size_t)TOK * 2 * 4;      // f32  [TOK][2]
    constexpr size_t OFF_PF   = OFF_TW   + (size_t)TOK * 2 * 4;      // f32  [64][E]
    constexpr size_t OFF_PC   = OFF_PF   + 64 * EE * 4;              // f32  [64][E]
    constexpr size_t OFF_TOFF = OFF_PC   + 64 * EE * 4;              // int  [E+1] (+pad)
    constexpr size_t OFF_LIST = OFF_TOFF + 256;                      // int  [E][CAP]

    __bf16* xb   = (__bf16*)(ws + OFF_XB);
    __bf16* w1t  = (__bf16*)(ws + OFF_W1T);
    __bf16* w2t  = (__bf16*)(ws + OFF_W2T);
    int*    tidx = (int*)   (ws + OFF_TIDX);
    float*  tw   = (float*) (ws + OFF_TW);
    float*  pf   = (float*) (ws + OFF_PF);
    float*  pc   = (float*) (ws + OFF_PC);
    int*    toff = (int*)   (ws + OFF_TOFF);
    int*    list = (int*)   (ws + OFF_LIST);

    const size_t nout = (size_t)out_size;  // TOK*D moe_out + 1 loss scalar
    zero_kernel<<<(unsigned)((nout + 255) / 256), 256, 0, stream>>>(out, nout);

    conv_x_kernel<<<(TOK * DD) / (256 * 8), 256, 0, stream>>>(x, xb);
    transpose_w_kernel<<<dim3(DD / 32, HH / 32, EE), 256, 0, stream>>>(w1, w1t, DD, HH);
    transpose_w_kernel<<<dim3(HH / 32, DD / 32, EE), 256, 0, stream>>>(w2, w2t, HH, DD);

    gate_kernel<<<TOK / 256, 256, 0, stream>>>(x, gw, gb, tidx, tw, pf, pc);
    route_kernel<<<EE, 256, 0, stream>>>(tidx, list);
    finalize_kernel<<<1, 32, 0, stream>>>(pf, pc, toff, out + (size_t)TOK * DD);

    ffn_kernel<<<FFN_BLOCKS, 256, 0, stream>>>(xb, w1t, w2t, b1, b2,
                                               tidx, tw, toff, list, out);
}
